// EBT_GAU_IN_41394894799308
// MI455X (gfx1250) — matrix operand order repaired and checked by reading, not compiled, not run on hardware
//
#include <hip/hip_runtime.h>

typedef __attribute__((ext_vector_type(16))) __bf16 v16bf;
typedef __attribute__((ext_vector_type(8)))  __bf16 v8bf;
typedef __attribute__((ext_vector_type(8)))  float  v8f;

#define NEGV (-1000000000.0f)

__device__ __forceinline__ float gelu_f(float x) {
  const float c0 = 0.7978845608028654f;
  return 0.5f * x * (1.0f + tanhf(c0 * (x + 0.044715f * x * x * x)));
}

// ---------------------------------------------------------------------------
// fp32 -> bf16 elementwise convert
// ---------------------------------------------------------------------------
__global__ void f2bf_kernel(const float* __restrict__ src, __bf16* __restrict__ dst, int n) {
  int i = blockIdx.x * blockDim.x + threadIdx.x;
  if (i < n) dst[i] = (__bf16)src[i];
}

// ---------------------------------------------------------------------------
// Pack a fp32 K x N weight matrix into WMMA B-fragment layout (bf16).
// Tile (ks, nt): 32 lanes, lane -> (half = lane>>4, r = lane&15).
// B frag element i of lane: B[(ks*32 + half*16 + i)*N + nt*16 + r].
// Packed: dst[((ks*nTiles + nt)*32 + lane)*16 + i]  -> per-lane 32B contiguous.
// ---------------------------------------------------------------------------
__global__ void pack_b_kernel(const float* __restrict__ src, __bf16* __restrict__ dst,
                              int K, int N) {
  int ks = blockIdx.x, nt = blockIdx.y, lane = threadIdx.x;
  int r16 = lane & 15, half = lane >> 4;
  int nTiles = N >> 4;
  __bf16* out = dst + ((size_t)(ks * nTiles + nt) * 32 + lane) * 16;
  const float* s = src + (size_t)(ks * 32 + half * 16) * N + nt * 16 + r16;
#pragma unroll
  for (int i = 0; i < 16; ++i) out[i] = (__bf16)s[(size_t)i * N];
}

// ---------------------------------------------------------------------------
// Generic WMMA GEMM:  Out[M,N] = A[M,K](bf16,row-major) * Bpacked + bias
// Block: 256 threads = 8 waves (2 x 4).  Block tile 64(M) x 256(N).
// Each wave: 2 A-frags x 4 B-frags -> 8 WMMAs per K-step (B reused 2x).
// EPI 0: fp32 out.  EPI 1: gelu -> bf16 out.
// ---------------------------------------------------------------------------
template <int EPI>
__global__ void gemm_wmma_kernel(const __bf16* __restrict__ A, const __bf16* __restrict__ Bp,
                                 const float* __restrict__ bias, void* __restrict__ Out,
                                 int M, int K, int N) {
  __shared__ __align__(16) __bf16 ldsA[64 * 32];
  const int tid  = threadIdx.x;
  const int wave = tid >> 5, lane = tid & 31;
  const int wm = wave >> 2, wn = wave & 3;
  const int r16 = lane & 15, half = lane >> 4;
  const int rowBase = blockIdx.y * 64;
  const int colBase = blockIdx.x * 256;
  const int nTiles = N >> 4;

  v8f acc[2][4];
#pragma unroll
  for (int u = 0; u < 2; ++u)
#pragma unroll
    for (int c = 0; c < 4; ++c)
#pragma unroll
      for (int j = 0; j < 8; ++j) acc[u][c][j] = 0.0f;

  const int idx = tid * 8;           // 2048 halves over 256 threads
  const int ar = idx >> 5, ac = idx & 31;

  for (int k0 = 0; k0 < K; k0 += 32) {
    // Stage 64x32 A tile into LDS (coalesced 16B per thread).
    const __bf16* gA = A + (size_t)(rowBase + ar) * K + k0 + ac;
    *(uint4*)&ldsA[idx] = *(const uint4*)gA;
    if (k0 + 32 < K) __builtin_prefetch((const void*)(gA + 32), 0, 1);
    __syncthreads();

    // Build two A fragments (16x32 each) from LDS: two b128 loads per lane.
    v16bf af[2];
#pragma unroll
    for (int u = 0; u < 2; ++u) {
      const __bf16* arow = &ldsA[(wm * 32 + u * 16 + r16) * 32 + half * 16];
      v8bf alo = *(const v8bf*)arow;
      v8bf ahi = *(const v8bf*)(arow + 8);
#pragma unroll
      for (int i = 0; i < 8; ++i) { af[u][i] = alo[i]; af[u][i + 8] = ahi[i]; }
    }

#pragma unroll
    for (int c = 0; c < 4; ++c) {
      int ct = (colBase >> 4) + wn * 4 + c;
      const __bf16* bptr = Bp + ((size_t)((k0 >> 5) * nTiles + ct) * 32 + lane) * 16;
      v8bf blo = *(const v8bf*)bptr;
      v8bf bhi = *(const v8bf*)(bptr + 8);
      v16bf bf;
#pragma unroll
      for (int i = 0; i < 8; ++i) { bf[i] = blo[i]; bf[i + 8] = bhi[i]; }
#pragma unroll
      for (int u = 0; u < 2; ++u)
        acc[u][c] = __builtin_amdgcn_wmma_f32_16x16x32_bf16(
            false, af[u], false, bf, (short)0, acc[u][c], false, false);
    }
    __syncthreads();
  }

#pragma unroll
  for (int u = 0; u < 2; ++u)
#pragma unroll
    for (int c = 0; c < 4; ++c) {
      int col = colBase + wn * 64 + c * 16 + r16;
      float bv = bias[col];
#pragma unroll
      for (int j = 0; j < 8; ++j) {
        int row = rowBase + wm * 32 + u * 16 + half * 8 + j;
        float v = acc[u][c][j] + bv;
        if (EPI == 1) {
          ((__bf16*)Out)[(size_t)row * N + col] = (__bf16)gelu_f(v);
        } else {
          ((float*)Out)[(size_t)row * N + col] = v;
        }
      }
    }
}

// ---------------------------------------------------------------------------
// Init LayerNorm: cs[n,0,s,:] = LN(tmp[n*32+s,:]) * g + be
// ---------------------------------------------------------------------------
__global__ void ln_init_kernel(const float* __restrict__ tmp, const float* __restrict__ g,
                               const float* __restrict__ be, float* __restrict__ cs) {
  int row = blockIdx.x;           // n*32 + s
  int n = row >> 5, s = row & 31;
  int tid = threadIdx.x;          // 256
  const float* src = tmp + (size_t)row * 512;
  float v0 = src[tid], v1 = src[tid + 256];
  __shared__ float red[256];
  red[tid] = v0 + v1;
  __syncthreads();
  for (int st = 128; st > 0; st >>= 1) {
    if (tid < st) red[tid] += red[tid + st];
    __syncthreads();
  }
  float mu = red[0] * (1.0f / 512.0f);
  __syncthreads();
  float d0 = v0 - mu, d1 = v1 - mu;
  red[tid] = d0 * d0 + d1 * d1;
  __syncthreads();
  for (int st = 128; st > 0; st >>= 1) {
    if (tid < st) red[tid] += red[tid + st];
    __syncthreads();
  }
  float rstd = rsqrtf(red[0] * (1.0f / 512.0f) + 1e-5f);
  float* dst = cs + ((size_t)n * 128 + s) * 512;   // beam 0 of (N,4,32,512)
  dst[tid]       = d0 * rstd * g[tid] + be[tid];
  dst[tid + 256] = d1 * rstd * g[tid + 256] + be[tid + 256];
}

// ---------------------------------------------------------------------------
// accu/bm init
// ---------------------------------------------------------------------------
__global__ void init_beam_kernel(float* accu, float* bm) {
  int i = blockIdx.x * blockDim.x + threadIdx.x;
  if (i < 1024) { accu[i] = 0.0f; bm[i] = 1.0f; }
}

// ---------------------------------------------------------------------------
// Composition scorer: comp[n,b,s] = Wd2 . gelu([l64 r64] @ Wd1 + bd1) + bd2
// Grid (S_loc, B_in, N), 64 threads.
// ---------------------------------------------------------------------------
__global__ void score_kernel(const float* __restrict__ cs, const float* __restrict__ Wd1,
                             const float* __restrict__ bd1, const float* __restrict__ Wd2,
                             const float* __restrict__ bd2, float* __restrict__ comp) {
  int s = blockIdx.x, b = blockIdx.y, n = blockIdx.z;
  int tid = threadIdx.x;  // 64
  __shared__ float sl[64], sr[64], red[64];
  const float* lrow = cs + ((size_t)(n * 4 + b) * 32 + s) * 512;
  sl[tid] = lrow[tid];
  sr[tid] = lrow[512 + tid];   // next position's row is contiguous
  __syncthreads();
  float acc = bd1[tid];
#pragma unroll 8
  for (int k = 0; k < 64; ++k) acc += sl[k] * Wd1[k * 64 + tid];
#pragma unroll 8
  for (int k = 0; k < 64; ++k) acc += sr[k] * Wd1[(64 + k) * 64 + tid];
  red[tid] = gelu_f(acc) * Wd2[tid];
  __syncthreads();
  for (int st = 32; st > 0; st >>= 1) {
    if (tid < st) red[tid] += red[tid + st];
    __syncthreads();
  }
  if (tid == 0) comp[(n * 4 + b) * 32 + s] = red[0] + bd2[0];
}

// ---------------------------------------------------------------------------
// Beam select: per sequence n, deterministic top-k per beam, expand, prune to 4.
// Writes accu/bm/parent/mpos for the 4 surviving beams.  One thread per n.
// ---------------------------------------------------------------------------
__global__ void beam_select_kernel(const float* __restrict__ comp,
                                   const float* __restrict__ imask,
                                   float* __restrict__ accu, float* __restrict__ bm,
                                   int* __restrict__ parent, int* __restrict__ mpos,
                                   int t, int S_loc, int B_in, int topk) {
  int n = threadIdx.x;  // 256
  float active = imask[n * 32 + t + 1];
  float cand_s[16], cand_b[16];
  int cand_p[16], cand_par[16];

  for (int b = 0; b < B_in; ++b) {
    float lg[32];
    float mx = -3.4e38f;
    for (int s = 0; s < S_loc; ++s) {
      float mk = imask[n * 32 + t + 1 + s];
      float v = (mk > 0.0f) ? comp[(n * 4 + b) * 32 + s] : NEGV;
      lg[s] = v;
      mx = fmaxf(mx, v);
    }
    float sum = 0.0f;
    for (int s = 0; s < S_loc; ++s) sum += expf(lg[s] - mx);
    bool used[32];
    for (int s = 0; s < S_loc; ++s) used[s] = false;
    float a0 = accu[n * 4 + b], b0 = bm[n * 4 + b];
    for (int kk = 0; kk < topk; ++kk) {
      int best = 0; float bv = -3.4e38f;
      for (int s = 0; s < S_loc; ++s)
        if (!used[s] && lg[s] > bv) { bv = lg[s]; best = s; }
      used[best] = true;
      float soft = expf(lg[best] - mx) / sum;
      float ns = active * logf(soft + 1e-20f);
      float nb = active + (1.0f - active) * ((kk == 0) ? 1.0f : 0.0f);
      int ci = b * topk + kk;
      cand_s[ci] = a0 + ns;
      cand_b[ci] = b0 * nb;
      cand_p[ci] = best;
      cand_par[ci] = b;
    }
  }

  int nc = B_in * topk;
  int order[4];
  if (nc > 4) {
    bool usedc[16];
    for (int i = 0; i < nc; ++i) usedc[i] = false;
    for (int kk = 0; kk < 4; ++kk) {
      int best = 0; float bv = -3.4e38f;
      for (int ci = 0; ci < nc; ++ci) {
        float v = (cand_b[ci] > 0.0f) ? cand_s[ci] : NEGV;
        if (!usedc[ci] && v > bv) { bv = v; best = ci; }
      }
      usedc[best] = true;
      order[kk] = best;
    }
  } else {
    for (int kk = 0; kk < 4; ++kk) order[kk] = kk % nc;
  }

  for (int kk = 0; kk < 4; ++kk) {
    int ci = order[kk];
    accu[n * 4 + kk] = cand_s[ci];
    bm[n * 4 + kk] = cand_b[ci];
    parent[n * 4 + kk] = cand_par[ci];
    mpos[n * 4 + kk] = cand_p[ci];
  }
}

// ---------------------------------------------------------------------------
// Gather [l | r] (bf16) for the GRC GEMM input.  Grid N*4, 256 threads.
// ---------------------------------------------------------------------------
__global__ void gather_lr_kernel(const float* __restrict__ cs, const int* __restrict__ parent,
                                 const int* __restrict__ mpos, __bf16* __restrict__ lr) {
  int row = blockIdx.x;
  int n = row >> 2, k = row & 3;
  int b = parent[n * 4 + k], pos = mpos[n * 4 + k];
  const float* lrow = cs + ((size_t)(n * 4 + b) * 32 + pos) * 512;  // r row follows
  __bf16* out = lr + (size_t)row * 1024;
  int tid = threadIdx.x;
#pragma unroll
  for (int q = 0; q < 4; ++q) {
    int j = tid + q * 256;
    out[j] = (__bf16)lrow[j];
  }
}

// ---------------------------------------------------------------------------
// GRC gate + LayerNorm: c(row,2048) viewed as [4][512]; gate-softmax over
// first 3; parent = g0*l + g1*r + g2*tanh(c3); LN with g_grc/be_grc.
// ---------------------------------------------------------------------------
__global__ void grc_gate_kernel(const float* __restrict__ c, const float* __restrict__ cs,
                                const int* __restrict__ parent, const int* __restrict__ mpos,
                                const float* __restrict__ g_grc, const float* __restrict__ be_grc,
                                float* __restrict__ node) {
  int row = blockIdx.x;
  int n = row >> 2, k = row & 3;
  int b = parent[n * 4 + k], pos = mpos[n * 4 + k];
  const float* crow = c + (size_t)row * 2048;
  const float* lrow = cs + ((size_t)(n * 4 + b) * 32 + pos) * 512;
  int tid = threadIdx.x;  // 256
  float pv[2];
#pragma unroll
  for (int q = 0; q < 2; ++q) {
    int d = tid + q * 256;
    float c0 = crow[d], c1 = crow[512 + d], c2 = crow[1024 + d], c3 = crow[1536 + d];
    float mx = fmaxf(c0, fmaxf(c1, c2));
    float e0 = expf(c0 - mx), e1 = expf(c1 - mx), e2 = expf(c2 - mx);
    float inv = 1.0f / (e0 + e1 + e2);
    pv[q] = (e0 * lrow[d] + e1 * lrow[512 + d] + e2 * tanhf(c3)) * inv;
  }
  __shared__ float red[256];
  red[tid] = pv[0] + pv[1];
  __syncthreads();
  for (int st = 128; st > 0; st >>= 1) {
    if (tid < st) red[tid] += red[tid + st];
    __syncthreads();
  }
  float mu = red[0] * (1.0f / 512.0f);
  __syncthreads();
  float d0 = pv[0] - mu, d1 = pv[1] - mu;
  red[tid] = d0 * d0 + d1 * d1;
  __syncthreads();
  for (int st = 128; st > 0; st >>= 1) {
    if (tid < st) red[tid] += red[tid + st];
    __syncthreads();
  }
  float rstd = rsqrtf(red[0] * (1.0f / 512.0f) + 1e-5f);
  node[(size_t)row * 512 + tid]       = d0 * rstd * g_grc[tid] + be_grc[tid];
  node[(size_t)row * 512 + tid + 256] = d1 * rstd * g_grc[tid + 256] + be_grc[tid + 256];
}

// ---------------------------------------------------------------------------
// Beam update: shift/insert into csNew for the 4 surviving beams.
// Grid (S_loc, 4, N), 128 threads (float4 per thread).
// ---------------------------------------------------------------------------
__global__ void update_cs_kernel(const float* __restrict__ csOld, const float* __restrict__ node,
                                 const int* __restrict__ parent, const int* __restrict__ mpos,
                                 const float* __restrict__ imask, float* __restrict__ csNew,
                                 int t) {
  int s = blockIdx.x, k = blockIdx.y, n = blockIdx.z;
  int b = parent[n * 4 + k], pos = mpos[n * 4 + k];
  float m = imask[n * 32 + t + 1];
  const float* src;
  if (m > 0.0f) {
    if (s < pos)       src = csOld + ((size_t)(n * 4 + b) * 32 + s) * 512;
    else if (s == pos) src = node + (size_t)(n * 4 + k) * 512;
    else               src = csOld + ((size_t)(n * 4 + b) * 32 + s + 1) * 512;
  } else {
    src = csOld + ((size_t)(n * 4 + b) * 32 + s) * 512;
  }
  float* dst = csNew + ((size_t)(n * 4 + k) * 32 + s) * 512;
  int tid = threadIdx.x;  // 128
  ((float4*)dst)[tid] = ((const float4*)src)[tid];
}

// ---------------------------------------------------------------------------
// Beam-probability pooling of roots -> pooled bf16.  Grid N, 128 threads.
// ---------------------------------------------------------------------------
__global__ void pool_kernel(const float* __restrict__ cs, const float* __restrict__ accu,
                            const float* __restrict__ bm, __bf16* __restrict__ pooled) {
  int n = blockIdx.x;
  int tid = threadIdx.x;  // 128
  float a[4], p[4];
  float mx = -3.4e38f;
#pragma unroll
  for (int k = 0; k < 4; ++k) {
    float v = (bm[n * 4 + k] > 0.0f) ? accu[n * 4 + k] : NEGV;
    a[k] = v;
    mx = fmaxf(mx, v);
  }
  float sum = 0.0f;
#pragma unroll
  for (int k = 0; k < 4; ++k) { p[k] = expf(a[k] - mx); sum += p[k]; }
  float inv = 1.0f / sum;
#pragma unroll
  for (int q = 0; q < 4; ++q) {
    int d = tid + q * 128;
    float acc = 0.0f;
#pragma unroll
    for (int k = 0; k < 4; ++k)
      acc += p[k] * inv * cs[((size_t)(n * 4 + k) * 32) * 512 + d];
    pooled[(size_t)n * 512 + d] = (__bf16)acc;
  }
}

// ---------------------------------------------------------------------------
// Host driver
// ---------------------------------------------------------------------------
extern "C" void kernel_launch(void* const* d_in, const int* in_sizes, int n_in,
                              void* d_out, int out_size, void* d_ws, size_t ws_size,
                              hipStream_t stream) {
  const int N = 256, S = 32, D = 512;

  const float* x      = (const float*)d_in[0];
  const float* imask  = (const float*)d_in[1];
  const float* W_init = (const float*)d_in[2];
  const float* b_init = (const float*)d_in[3];
  const float* g_init = (const float*)d_in[4];
  const float* be_init= (const float*)d_in[5];
  const float* Wd1    = (const float*)d_in[6];
  const float* bd1    = (const float*)d_in[7];
  const float* Wd2    = (const float*)d_in[8];
  const float* bd2    = (const float*)d_in[9];
  const float* Wc1    = (const float*)d_in[10];
  const float* bc1    = (const float*)d_in[11];
  const float* Wc2    = (const float*)d_in[12];
  const float* bc2    = (const float*)d_in[13];
  const float* g_grc  = (const float*)d_in[14];
  const float* be_grc = (const float*)d_in[15];
  const float* Wt1    = (const float*)d_in[16];
  const float* bt1    = (const float*)d_in[17];
  const float* Wt2    = (const float*)d_in[18];
  const float* bt2    = (const float*)d_in[19];

  char* w = (char*)d_ws;
  auto take = [&](size_t bytes) -> char* {
    char* r = w;
    w += (bytes + 255) & ~(size_t)255;
    return r;
  };

  float*  csA     = (float*)take((size_t)N * 4 * S * D * 4);      // 64 MB
  float*  csB     = (float*)take((size_t)N * 4 * S * D * 4);      // 64 MB
  __bf16* x_bf    = (__bf16*)take((size_t)N * S * D * 2);         // 8 MB
  __bf16* Wi_pk   = (__bf16*)take((size_t)512 * 512 * 2);
  __bf16* Wc1_pk  = (__bf16*)take((size_t)1024 * 2048 * 2);
  __bf16* Wc2_pk  = (__bf16*)take((size_t)2048 * 2048 * 2);
  __bf16* Wt1_pk  = (__bf16*)take((size_t)512 * 512 * 2);
  __bf16* Wt2_pk  = (__bf16*)take((size_t)512 * 512 * 2);
  float*  tmpI    = (float*)take((size_t)N * S * D * 4);          // 16 MB
  __bf16* lr_bf   = (__bf16*)take((size_t)1024 * 1024 * 2);
  __bf16* h_bf    = (__bf16*)take((size_t)1024 * 2048 * 2);
  float*  c_f     = (float*)take((size_t)1024 * 2048 * 4);
  float*  node    = (float*)take((size_t)1024 * 512 * 4);
  float*  comp    = (float*)take((size_t)N * 4 * 32 * 4);
  float*  accu    = (float*)take((size_t)N * 4 * 4);
  float*  bm      = (float*)take((size_t)N * 4 * 4);
  int*    parent  = (int*)take((size_t)N * 4 * 4);
  int*    mpos    = (int*)take((size_t)N * 4 * 4);
  __bf16* pool_bf = (__bf16*)take((size_t)N * 512 * 2);
  __bf16* ht_bf   = (__bf16*)take((size_t)N * 512 * 2);

  // --- weight packing + input conversion ---
  {
    int n = N * S * D;
    f2bf_kernel<<<(n + 255) / 256, 256, 0, stream>>>(x, x_bf, n);
  }
  pack_b_kernel<<<dim3(512 / 32, 512 / 16), 32, 0, stream>>>(W_init, Wi_pk, 512, 512);
  pack_b_kernel<<<dim3(1024 / 32, 2048 / 16), 32, 0, stream>>>(Wc1, Wc1_pk, 1024, 2048);
  pack_b_kernel<<<dim3(2048 / 32, 2048 / 16), 32, 0, stream>>>(Wc2, Wc2_pk, 2048, 2048);
  pack_b_kernel<<<dim3(512 / 32, 512 / 16), 32, 0, stream>>>(Wt1, Wt1_pk, 512, 512);
  pack_b_kernel<<<dim3(512 / 32, 512 / 16), 32, 0, stream>>>(Wt2, Wt2_pk, 512, 512);

  // --- init: cs0 = LN(x @ W_init + b_init) ---
  gemm_wmma_kernel<0><<<dim3(512 / 256, (N * S) / 64), 256, 0, stream>>>(
      x_bf, Wi_pk, b_init, tmpI, N * S, 512, 512);
  ln_init_kernel<<<N * S, 256, 0, stream>>>(tmpI, g_init, be_init, csA);
  init_beam_kernel<<<4, 256, 0, stream>>>(accu, bm);

  // --- composition loop ---
  float* csCur = csA;
  float* csNxt = csB;
  for (int t = 0; t < S - 1; ++t) {
    int Sp1 = S - t;
    int S_loc = Sp1 - 1;
    int B_in = (t == 0) ? 1 : 4;
    int topk = (S_loc < 4) ? S_loc : 4;

    score_kernel<<<dim3(S_loc, B_in, N), 64, 0, stream>>>(csCur, Wd1, bd1, Wd2, bd2, comp);
    beam_select_kernel<<<1, 256, 0, stream>>>(comp, imask, accu, bm, parent, mpos,
                                              t, S_loc, B_in, topk);
    gather_lr_kernel<<<N * 4, 256, 0, stream>>>(csCur, parent, mpos, lr_bf);
    gemm_wmma_kernel<1><<<dim3(2048 / 256, 1024 / 64), 256, 0, stream>>>(
        lr_bf, Wc1_pk, bc1, h_bf, 1024, 1024, 2048);
    gemm_wmma_kernel<0><<<dim3(2048 / 256, 1024 / 64), 256, 0, stream>>>(
        h_bf, Wc2_pk, bc2, c_f, 1024, 2048, 2048);
    grc_gate_kernel<<<N * 4, 256, 0, stream>>>(c_f, csCur, parent, mpos, g_grc, be_grc, node);
    update_cs_kernel<<<dim3(S_loc, 4, N), 128, 0, stream>>>(csCur, node, parent, mpos,
                                                            imask, csNxt, t);
    float* tswap = csCur; csCur = csNxt; csNxt = tswap;
  }

  // --- head ---
  pool_kernel<<<N, 128, 0, stream>>>(csCur, accu, bm, pool_bf);
  gemm_wmma_kernel<1><<<dim3(512 / 256, N / 64), 256, 0, stream>>>(
      pool_bf, Wt1_pk, bt1, ht_bf, N, 512, 512);
  gemm_wmma_kernel<0><<<dim3(512 / 256, N / 64), 256, 0, stream>>>(
      ht_bf, Wt2_pk, bt2, (float*)d_out, N, 512, 512);
}
